// MambaGF_40355512713857
// MI455X (gfx1250) — compile-verified, matrix operand order checked by reading, not run on hardware
//
#include <hip/hip_runtime.h>
#include <hip/hip_bf16.h>

// MambaGF for gfx1250 (MI455X). Dense layers: v_wmma_f32_16x16x32_f16 with
// fragment-ready LDS tiles. Selective scan: chunked parallel scan (pass1 local
// states + decay products, sequential combine, pass2 emit) with async
// global->LDS staging of the x_dbl slab per chunk.

typedef __attribute__((ext_vector_type(16))) _Float16 v16h;
typedef __attribute__((ext_vector_type(8)))  _Float16 v8h;
typedef __attribute__((ext_vector_type(4)))  _Float16 v4h;
typedef __attribute__((ext_vector_type(8)))  float    v8f;
typedef int v4i_ __attribute__((vector_size(16)));   // matches async-lds builtin

#define BSZ   8
#define LL    4096      // 64*64
#define BL    32768     // BSZ*LL
#define INC   96
#define DIMC  128
#define DI_   256
#define NST   16
#define RR    8
#define NCH   32        // scan chunks
#define CLEN  128       // steps per chunk

#if defined(__has_builtin)
# if __has_builtin(__builtin_amdgcn_global_load_async_to_lds_b128)
#  define HAVE_ASYNC_LDS 1
# endif
#endif

// scan-direction permutation (involution): maps scan index -> spatial index
__device__ __forceinline__ int permk(int k, int j) {
  switch (k & 3) {
    case 0: return j;
    case 1: return ((j & 63) << 6) | (j >> 6);
    case 2: return 4095 - j;
    default: { int t = 4095 - j; return ((t & 63) << 6) | (t >> 6); }
  }
}

__device__ __forceinline__ float sigmoidf_(float x) { return 1.f / (1.f + __expf(-x)); }

// CDNA5 16-bit 16x32 fragment layout: lane half h (0:lanes0-15,1:lanes16-31)
// holds K = {8h..8h+7, 16+8h..16+8h+7} as elements 0..15.
__device__ __forceinline__ int fragpos(int K) {
  return (((K >> 3) & 1) << 4) | ((K & 7) | ((K & 16) >> 1));
}

// ---------------------------------------------------------------------------
// WMMA GEMM: out = act(A @ W^T + bias) [+ res]      (see prior rounds)
// ---------------------------------------------------------------------------
template<int MODEA, int MODEOUT, int ACT>
__global__ __launch_bounds__(256) void gemm_wmma(
    const float* __restrict__ A, const float* __restrict__ W,
    const float* __restrict__ bias, const float* __restrict__ res,
    float* __restrict__ out, const float* __restrict__ gsrc,
    int M, int Nd, int Kd)
{
  __shared__ _Float16 lA[64][40];   // fragment-ready, 80B row stride
  __shared__ _Float16 lB[64][40];

  const int tid  = threadIdx.x;
  const int lane = tid & 31;
  const int wid  = tid >> 5;
  const int wm   = wid >> 1;
  const int wn   = wid & 1;
  const int rowBase = blockIdx.x * 64;
  const int colBase = blockIdx.y * 64;

  const float* Wp = W;
  if (MODEA == 2) Wp = W + (size_t)((rowBase >> 12) & 3) * Nd * Kd;

  v8f acc0 = {}; v8f acc1 = {};
  const int nK = Kd >> 5;

  for (int kt = 0; kt < nK; ++kt) {
    const int k0 = kt << 5;
    #pragma unroll
    for (int i = 0; i < 2; ++i) {     // A tile 64x32 as 512 K-quads
      int q = tid + (i << 8);
      int r = q >> 3, kq = (q & 7) << 2;
      int grow = rowBase + r, gk = k0 + kq;
      float4 v;
      if (MODEA == 0) {
        v = *(const float4*)(A + (size_t)grow * Kd + gk);
      } else if (MODEA == 1) {
        int b = grow >> 12, l = grow & 4095;
        const float* g0 = gsrc + ((size_t)b * Kd + gk) * LL + l;
        v.x = g0[0]; v.y = g0[LL]; v.z = g0[2 * LL]; v.w = g0[3 * LL];
      } else {
        int b = grow >> 14, kk = (grow >> 12) & 3, j = grow & 4095;
        const float* g0 = gsrc + ((size_t)b * 256 + gk) * LL + permk(kk, j);
        v.x = g0[0]; v.y = g0[LL]; v.z = g0[2 * LL]; v.w = g0[3 * LL];
      }
      v4h h4;
      h4[0] = (_Float16)v.x; h4[1] = (_Float16)v.y;
      h4[2] = (_Float16)v.z; h4[3] = (_Float16)v.w;
      *(v4h*)&lA[r][fragpos(kq)] = h4;
    }
    #pragma unroll
    for (int i = 0; i < 2; ++i) {     // B tile 32x64 (fragment-ready)
      int q = tid + (i << 8);
      int n = q >> 3, kq = (q & 7) << 2;
      int gn = colBase + n, gk = k0 + kq;
      float4 v = {0.f, 0.f, 0.f, 0.f};
      if (gn < Nd) v = *(const float4*)(Wp + (size_t)gn * Kd + gk);
      v4h h4;
      h4[0] = (_Float16)v.x; h4[1] = (_Float16)v.y;
      h4[2] = (_Float16)v.z; h4[3] = (_Float16)v.w;
      *(v4h*)&lB[n][fragpos(kq)] = h4;
    }
    __syncthreads();

    const int mrow  = wm * 16 + (lane & 15);
    const int hb    = (lane >> 4) << 4;
    const int ncol0 = wn * 32 + (lane & 15);
    v8h a_lo = *(const v8h*)&lA[mrow][hb];
    v8h a_hi = *(const v8h*)&lA[mrow][hb + 8];
    v8h b0lo = *(const v8h*)&lB[ncol0][hb];
    v8h b0hi = *(const v8h*)&lB[ncol0][hb + 8];
    v8h b1lo = *(const v8h*)&lB[ncol0 + 16][hb];
    v8h b1hi = *(const v8h*)&lB[ncol0 + 16][hb + 8];
    v16h af, bf0, bf1;
    #pragma unroll
    for (int e = 0; e < 8; ++e) {
      af[e]  = a_lo[e]; af[8 + e]  = a_hi[e];
      bf0[e] = b0lo[e]; bf0[8 + e] = b0hi[e];
      bf1[e] = b1lo[e]; bf1[8 + e] = b1hi[e];
    }
    acc0 = __builtin_amdgcn_wmma_f32_16x16x32_f16(false, af, false, bf0,
                                                  (short)0, acc0, false, false);
    acc1 = __builtin_amdgcn_wmma_f32_16x16x32_f16(false, af, false, bf1,
                                                  (short)0, acc1, false, false);
    __syncthreads();
  }

  const int rhalf = (lane >> 4) << 3;
  #pragma unroll
  for (int g = 0; g < 8; ++g) {
    int grow = rowBase + wm * 16 + g + rhalf;
    #pragma unroll
    for (int h = 0; h < 2; ++h) {
      int c = colBase + wn * 32 + h * 16 + (lane & 15);
      if (c >= Nd) continue;
      float v = h ? acc1[g] : acc0[g];
      if (bias) v += bias[c];
      if (ACT == 1) {
        float t = v;
        v = 0.5f * t * (1.f + tanhf(0.7978845608f * (t + 0.044715f * t * t * t)));
      }
      if (res) v += res[(size_t)grow * Nd + c];
      if (MODEOUT == 0) {
        out[(size_t)grow * Nd + c] = v;
      } else if (MODEOUT == 1) {
        int b = grow >> 12, l = grow & 4095;
        out[((size_t)b * Nd + c) * LL + l] = v;
      } else {
        int b = grow >> 14, kk = (grow >> 12) & 3, j = grow & 4095;
        out[(((size_t)(b * 4 + kk)) * 40 + c) * LL + j] = v;
      }
    }
  }
}

// ---------------------------------------------------------------------------
// LayerNorm (one wave32 per row) with optional silu(z) gate.
// ---------------------------------------------------------------------------
__global__ __launch_bounds__(256) void ln_kernel(
    const float* __restrict__ x, const float* __restrict__ gm,
    const float* __restrict__ bt, const float* __restrict__ z, int zStride,
    float* __restrict__ out, int M, int C)
{
  int wid = threadIdx.x >> 5, lane = threadIdx.x & 31;
  int row = blockIdx.x * 8 + wid;
  if (row >= M) return;
  const float* xr = x + (size_t)row * C;
  const int nper = C >> 5;
  float vb[8];
  float s = 0.f;
  #pragma unroll
  for (int i = 0; i < 8; ++i)
    if (i < nper) { vb[i] = xr[lane + (i << 5)]; s += vb[i]; }
  #pragma unroll
  for (int o = 16; o; o >>= 1) s += __shfl_xor(s, o, 32);
  float mean = s / (float)C;
  float q = 0.f;
  #pragma unroll
  for (int i = 0; i < 8; ++i)
    if (i < nper) { float d = vb[i] - mean; q += d * d; }
  #pragma unroll
  for (int o = 16; o; o >>= 1) q += __shfl_xor(q, o, 32);
  float inv = rsqrtf(q / (float)C + 1e-5f);
  float* orow = out + (size_t)row * C;
  #pragma unroll
  for (int i = 0; i < 8; ++i) {
    if (i < nper) {
      int c = lane + (i << 5);
      float v = (vb[i] - mean) * inv * gm[c] + bt[c];
      if (z) { float zz = z[(size_t)row * zStride + c]; v *= zz * sigmoidf_(zz); }
      orow[c] = v;
    }
  }
}

// ---------------------------------------------------------------------------
// Depthwise 3x3 conv (SAME) + bias + SiLU: xz (B,L,512)[:256] -> xc (B,256,L)
// ---------------------------------------------------------------------------
__global__ __launch_bounds__(256) void dwconv_silu(
    const float* __restrict__ xz, const float* __restrict__ cw,
    const float* __restrict__ cb, float* __restrict__ xc)
{
  int pix = blockIdx.x;
  int b = pix >> 12, l = pix & 4095;
  int h = l >> 6, w = l & 63;
  int d = threadIdx.x;
  float s = 0.f;
  #pragma unroll
  for (int kh = 0; kh < 3; ++kh) {
    int hh = h + kh - 1;
    if ((unsigned)hh >= 64u) continue;
    #pragma unroll
    for (int kw = 0; kw < 3; ++kw) {
      int ww = w + kw - 1;
      if ((unsigned)ww >= 64u) continue;
      s += xz[((size_t)((b << 12) | ((hh << 6) | ww))) * 512 + d]
           * cw[d * 9 + kh * 3 + kw];
    }
  }
  s += cb[d];
  s = s * sigmoidf_(s);
  xc[((size_t)b * 256 + d) * LL + l] = s;
}

// ---------------------------------------------------------------------------
// Stage NROWS rows x 128 steps of the x_dbl slab into LDS (async if available)
// ---------------------------------------------------------------------------
template<int NROWS>
__device__ __forceinline__ void stage_rows(const float* __restrict__ gbase,
                                           float* sd, int tid) {
  #pragma unroll
  for (int i = 0; i < (NROWS * 32) / 256; ++i) {
    int q = tid + (i << 8);
    int c = q >> 5, jj = (q & 31) << 2;
    const float* gp = gbase + (size_t)c * LL + jj;
    float* lp = sd + c * CLEN + jj;
#ifdef HAVE_ASYNC_LDS
    __builtin_amdgcn_global_load_async_to_lds_b128((v4i_*)gp, (v4i_*)lp, 0, 0);
#else
    *(float4*)lp = *(const float4*)gp;
#endif
  }
#ifdef HAVE_ASYNC_LDS
  asm volatile("s_wait_asynccnt 0x0" ::: "memory");
#endif
  __syncthreads();
}

// ---------------------------------------------------------------------------
// Chunked selective scan.
// Pass 1: per (b,k,chunk,d): local scan with h_in=0; store end state h_out and
//         decay product P = prod(exp(dt*A)).
// Combine: per (b,k,d): sequential over 32 chunks; rewrite hout[c] := h_in(c).
// Pass 2: per (b,k,chunk,d): rerun chunk from h_in, emit y (dt-proj + softplus
//         + recurrence + C-dot + D*u), cross-merge via atomicAdd.
// ---------------------------------------------------------------------------
__global__ __launch_bounds__(256) void scan_pass1(
    const float* __restrict__ xc, const float* __restrict__ xdbl,
    const float* __restrict__ dtw, const float* __restrict__ dtb,
    const float* __restrict__ Alog,
    float* __restrict__ hout, float* __restrict__ pout)
{
  __shared__ float sd[24 * CLEN];          // dt rows 0..7, B rows 8..23
  int bkc = blockIdx.x;                    // (b*4+k)*32 + chunk
  int bk = bkc >> 5, ch = bkc & 31;
  int b = bk >> 2, kk = bk & 3;
  int d = threadIdx.x;
  int cd = kk * 256 + d;

  const float* xdb = xdbl + ((size_t)bk * 40) * LL + ch * CLEN;
  stage_rows<24>(xdb, sd, d);

  float wdt[RR];
  #pragma unroll
  for (int r = 0; r < RR; ++r) wdt[r] = dtw[(size_t)cd * RR + r];
  float bdt = dtb[cd];
  float Af[NST];
  #pragma unroll
  for (int n = 0; n < NST; ++n) Af[n] = -__expf(Alog[(size_t)cd * NST + n]);
  const float* xcb = xc + ((size_t)b * 256 + d) * LL;

  float h[NST], P[NST];
  #pragma unroll
  for (int n = 0; n < NST; ++n) { h[n] = 0.f; P[n] = 1.f; }

  for (int jj = 0; jj < CLEN; ++jj) {
    int j = ch * CLEN + jj;
    float u = xcb[permk(kk, j)];
    float dtr = bdt;
    #pragma unroll
    for (int r = 0; r < RR; ++r) dtr += wdt[r] * sd[r * CLEN + jj];
    float dt = (dtr > 20.f) ? dtr : log1pf(__expf(dtr));
    float du = dt * u;
    #pragma unroll
    for (int n = 0; n < NST; ++n) {
      float a = __expf(dt * Af[n]);
      h[n] = a * h[n] + du * sd[(8 + n) * CLEN + jj];
      P[n] *= a;
    }
  }
  float* hp = hout + ((size_t)bkc * 256 + d) * NST;
  float* pp = pout + ((size_t)bkc * 256 + d) * NST;
  #pragma unroll
  for (int t = 0; t < 4; ++t) {
    float4 v; v.x = h[4*t]; v.y = h[4*t+1]; v.z = h[4*t+2]; v.w = h[4*t+3];
    *(float4*)(hp + 4 * t) = v;
    float4 w; w.x = P[4*t]; w.y = P[4*t+1]; w.z = P[4*t+2]; w.w = P[4*t+3];
    *(float4*)(pp + 4 * t) = w;
  }
}

__global__ __launch_bounds__(256) void scan_combine(
    float* __restrict__ hout, const float* __restrict__ pout)
{
  int bk = blockIdx.x, d = threadIdx.x;
  float hin[NST];
  #pragma unroll
  for (int n = 0; n < NST; ++n) hin[n] = 0.f;
  for (int c = 0; c < NCH; ++c) {
    size_t base = ((size_t)(bk * NCH + c) * 256 + d) * NST;
    float hl[NST], pl[NST];
    #pragma unroll
    for (int t = 0; t < 4; ++t) {
      float4 v = *(const float4*)(hout + base + 4 * t);
      hl[4*t] = v.x; hl[4*t+1] = v.y; hl[4*t+2] = v.z; hl[4*t+3] = v.w;
      float4 w = *(const float4*)(pout + base + 4 * t);
      pl[4*t] = w.x; pl[4*t+1] = w.y; pl[4*t+2] = w.z; pl[4*t+3] = w.w;
    }
    #pragma unroll
    for (int t = 0; t < 4; ++t) {       // hout[c] := h_in for chunk c
      float4 v; v.x = hin[4*t]; v.y = hin[4*t+1];
      v.z = hin[4*t+2]; v.w = hin[4*t+3];
      *(float4*)(hout + base + 4 * t) = v;
    }
    #pragma unroll
    for (int n = 0; n < NST; ++n) hin[n] = hl[n] + pl[n] * hin[n];
  }
}

__global__ __launch_bounds__(256) void scan_pass2(
    const float* __restrict__ xc, const float* __restrict__ xdbl,
    const float* __restrict__ dtw, const float* __restrict__ dtb,
    const float* __restrict__ Alog, const float* __restrict__ Ds,
    const float* __restrict__ hin_buf, float* __restrict__ ybuf)
{
  __shared__ float sd[40 * CLEN];          // dt 0..7, B 8..23, C 24..39
  int bkc = blockIdx.x;
  int bk = bkc >> 5, ch = bkc & 31;
  int b = bk >> 2, kk = bk & 3;
  int d = threadIdx.x;
  int cd = kk * 256 + d;

  const float* xdb = xdbl + ((size_t)bk * 40) * LL + ch * CLEN;
  stage_rows<40>(xdb, sd, d);

  float wdt[RR];
  #pragma unroll
  for (int r = 0; r < RR; ++r) wdt[r] = dtw[(size_t)cd * RR + r];
  float bdt = dtb[cd];
  float Af[NST];
  #pragma unroll
  for (int n = 0; n < NST; ++n) Af[n] = -__expf(Alog[(size_t)cd * NST + n]);
  float Dv = Ds[cd];
  const float* xcb = xc + ((size_t)b * 256 + d) * LL;

  float h[NST];
  const float* hp = hin_buf + ((size_t)bkc * 256 + d) * NST;
  #pragma unroll
  for (int t = 0; t < 4; ++t) {
    float4 v = *(const float4*)(hp + 4 * t);
    h[4*t] = v.x; h[4*t+1] = v.y; h[4*t+2] = v.z; h[4*t+3] = v.w;
  }
  float* yb = ybuf + ((size_t)b << 12) * 256 + d;

  for (int jj = 0; jj < CLEN; ++jj) {
    int j = ch * CLEN + jj;
    int pos = permk(kk, j);
    float u = xcb[pos];
    float dtr = bdt;
    #pragma unroll
    for (int r = 0; r < RR; ++r) dtr += wdt[r] * sd[r * CLEN + jj];
    float dt = (dtr > 20.f) ? dtr : log1pf(__expf(dtr));
    float du = dt * u;
    float y = 0.f;
    #pragma unroll
    for (int n = 0; n < NST; ++n) {
      h[n] = __expf(dt * Af[n]) * h[n] + du * sd[(8 + n) * CLEN + jj];
      y += h[n] * sd[(24 + n) * CLEN + jj];
    }
    y += u * Dv;
    atomicAdd(yb + (size_t)pos * 256, y);
  }
}

// ---------------------------------------------------------------------------
// BatchNorm stats + gating
// ---------------------------------------------------------------------------
__global__ __launch_bounds__(256) void bnstats_kernel(
    const float* __restrict__ t1, const float* __restrict__ t2,
    float* __restrict__ stats)
{
  int bi = blockIdx.x;
  const float* t = (bi < 128) ? t1 : t2;
  int c = bi & 127;
  __shared__ float ss[256], sq[256];
  float s = 0.f, q = 0.f;
  for (int i = threadIdx.x; i < BL; i += 256) {
    float v = t[(size_t)i * 128 + c];
    s += v; q += v * v;
  }
  ss[threadIdx.x] = s; sq[threadIdx.x] = q;
  __syncthreads();
  for (int o = 128; o; o >>= 1) {
    if (threadIdx.x < (unsigned)o) {
      ss[threadIdx.x] += ss[threadIdx.x + o];
      sq[threadIdx.x] += sq[threadIdx.x + o];
    }
    __syncthreads();
  }
  if (threadIdx.x == 0) { stats[bi] = ss[0]; stats[256 + bi] = sq[0]; }
}

__global__ __launch_bounds__(256) void gate_kernel(
    const float* __restrict__ t1, const float* __restrict__ t2,
    const float* __restrict__ stats,
    const float* __restrict__ g1, const float* __restrict__ b1,
    const float* __restrict__ g2, const float* __restrict__ b2,
    float* __restrict__ out)
{
  int idx = blockIdx.x * 256 + threadIdx.x;
  int c = idx & 127;
  const float Nc = (float)BL;
  float m1 = stats[c] / Nc,        v1 = stats[256 + c] / Nc - m1 * m1;
  float m2 = stats[128 + c] / Nc,  v2 = stats[384 + c] / Nc - m2 * m2;
  float a  = (t1[idx] - m1) * rsqrtf(v1 + 1e-5f) * g1[c] + b1[c];
  float bb = (t2[idx] - m2) * rsqrtf(v2 + 1e-5f) * g2[c] + b2[c];
  out[idx] = a * sigmoidf_(bb);
}

// ---------------------------------------------------------------------------
extern "C" void kernel_launch(void* const* d_in, const int* in_sizes, int n_in,
                              void* d_out, int out_size, void* d_ws, size_t ws_size,
                              hipStream_t stream) {
  (void)in_sizes; (void)n_in; (void)out_size; (void)ws_size;
  float* wsf = (float*)d_ws;

  float* X0   = wsf;                              // BL*128
  float* XLN  = X0   + (size_t)BL * 128;          // BL*128
  float* XZ   = XLN  + (size_t)BL * 128;          // BL*512
  float* XC   = XZ   + (size_t)BL * 512;          // B*256*L
  float* XDBL = XC   + (size_t)BSZ * 256 * LL;    // B*4*40*L
  float* YB   = XDBL + (size_t)BSZ * 4 * 40 * LL; // BL*256
  float* X1   = YB   + (size_t)BL * 256;          // BL*128
  float* T1   = X1   + (size_t)BL * 128;          // BL*128
  float* T2   = T1   + (size_t)BL * 128;          // BL*128
  float* STAT = T2   + (size_t)BL * 128;          // 512
  float* HOUT = STAT + 512;                       // 32*32*256*16
  float* POUT = HOUT + (size_t)32 * NCH * 256 * NST;

  dim3 blk(256);

  for (int br = 0; br < 2; ++br) {
    const float* feat = (const float*)d_in[br];
    int pb = 2 + 21 * br;
    const float* c1_w   = (const float*)d_in[pb + 0];
    const float* c1_b   = (const float*)d_in[pb + 1];
    const float* ln1_g  = (const float*)d_in[pb + 2];
    const float* ln1_b  = (const float*)d_in[pb + 3];
    const float* inp_w  = (const float*)d_in[pb + 4];
    const float* conv_w = (const float*)d_in[pb + 5];
    const float* conv_b = (const float*)d_in[pb + 6];
    const float* xprj_w = (const float*)d_in[pb + 7];
    const float* dt_w   = (const float*)d_in[pb + 8];
    const float* dt_b   = (const float*)d_in[pb + 9];
    const float* A_log  = (const float*)d_in[pb + 10];
    const float* Ds     = (const float*)d_in[pb + 11];
    const float* on_g   = (const float*)d_in[pb + 12];
    const float* on_b   = (const float*)d_in[pb + 13];
    const float* oprj_w = (const float*)d_in[pb + 14];
    const float* ln2_g  = (const float*)d_in[pb + 15];
    const float* ln2_b  = (const float*)d_in[pb + 16];
    const float* fc1_w  = (const float*)d_in[pb + 17];
    const float* fc1_b  = (const float*)d_in[pb + 18];
    const float* fc2_w  = (const float*)d_in[pb + 19];
    const float* fc2_b  = (const float*)d_in[pb + 20];
    float* T = br ? T2 : T1;

    // 1) c1 1x1 conv: gather (B,96,L) -> x0 (BL,128)
    gemm_wmma<1, 0, 0><<<dim3(512, 2), blk, 0, stream>>>(
        nullptr, c1_w, c1_b, nullptr, X0, feat, BL, DIMC, INC);
    // 2) ln1
    ln_kernel<<<BL / 8, blk, 0, stream>>>(X0, ln1_g, ln1_b, nullptr, 0, XLN, BL, DIMC);
    // 3) in_proj 128 -> 512 (xi | z)
    gemm_wmma<0, 0, 0><<<dim3(512, 8), blk, 0, stream>>>(
        XLN, inp_w, nullptr, nullptr, XZ, nullptr, BL, 512, DIMC);
    // 4) depthwise conv + silu -> xc (B,256,L)
    dwconv_silu<<<BL, blk, 0, stream>>>(XZ, conv_w, conv_b, XC);
    // 5) x_proj: cross-scan gather GEMM -> xdbl (B,4,40,L)
    gemm_wmma<2, 2, 0><<<dim3(2048, 1), blk, 0, stream>>>(
        nullptr, xprj_w, nullptr, nullptr, XDBL, XC, BL * 4, 40, DI_);
    // 6) chunked selective scan + cross-merge
    (void)hipMemsetAsync(YB, 0, (size_t)BL * 256 * sizeof(float), stream);
    scan_pass1<<<32 * NCH, blk, 0, stream>>>(XC, XDBL, dt_w, dt_b, A_log, HOUT, POUT);
    scan_combine<<<32, blk, 0, stream>>>(HOUT, POUT);
    scan_pass2<<<32 * NCH, blk, 0, stream>>>(XC, XDBL, dt_w, dt_b, A_log, Ds, HOUT, YB);
    // 7) out_norm LN * silu(z)  (in-place on YB; z = xz[...,256:])
    ln_kernel<<<BL / 8, blk, 0, stream>>>(YB, on_g, on_b, XZ + 256, 512, YB, BL, DI_);
    // 8) out_proj 256 -> 128, + residual x0 -> x1
    gemm_wmma<0, 0, 0><<<dim3(512, 2), blk, 0, stream>>>(
        YB, oprj_w, nullptr, X0, X1, nullptr, BL, DIMC, DI_);
    // 9) ln2
    ln_kernel<<<BL / 8, blk, 0, stream>>>(X1, ln2_g, ln2_b, nullptr, 0, XLN, BL, DIMC);
    // 10) fc1 + gelu (reuse XZ as hidden)
    gemm_wmma<0, 0, 1><<<dim3(512, 8), blk, 0, stream>>>(
        XLN, fc1_w, fc1_b, nullptr, XZ, nullptr, BL, 512, DIMC);
    // 11) fc2 + residual x1 -> branch output
    gemm_wmma<0, 0, 0><<<dim3(512, 2), blk, 0, stream>>>(
        XZ, fc2_w, fc2_b, X1, T, nullptr, BL, DIMC, 512);
  }

  const float* bn1_g = (const float*)d_in[44];
  const float* bn1_b = (const float*)d_in[45];
  const float* bn2_g = (const float*)d_in[46];
  const float* bn2_b = (const float*)d_in[47];
  const float* out_w = (const float*)d_in[48];
  const float* out_b = (const float*)d_in[49];

  bnstats_kernel<<<256, blk, 0, stream>>>(T1, T2, STAT);
  gate_kernel<<<(BL * 128) / 256, blk, 0, stream>>>(
      T1, T2, STAT, bn1_g, bn1_b, bn2_g, bn2_b, X0);
  // final 1x1: (BL,128) -> (B,192,L) channel-major into d_out
  gemm_wmma<0, 1, 0><<<dim3(512, 3), blk, 0, stream>>>(
      X0, out_w, out_b, nullptr, (float*)d_out, nullptr, BL, 2 * INC, DIMC);
}
